// Attention_Encode_1821066134150
// MI455X (gfx1250) — matrix operand order repaired and checked by reading, not compiled, not run on hardware
//
#include <hip/hip_runtime.h>

typedef __attribute__((ext_vector_type(16))) _Float16 v16h;
typedef __attribute__((ext_vector_type(8)))  float    v8f;
typedef __attribute__((ext_vector_type(4)))  unsigned u32x4;
typedef __attribute__((ext_vector_type(8)))  int      i32x8;
typedef __attribute__((ext_vector_type(4)))  int      i32x4;

#define LDG_PAD 40   // GEMM LDS row stride in halves (80B, 16B-aligned rows)
#define LDF_PAD 72   // flash LDS row stride in halves (144B, 16B-aligned rows)

#if __has_builtin(__builtin_amdgcn_tensor_load_to_lds)
#define ATHENA_TDM 1
#else
#define ATHENA_TDM 0
#endif

#if __has_builtin(__builtin_amdgcn_s_wait_tensorcnt)
#define WAIT_TENSOR(n) __builtin_amdgcn_s_wait_tensorcnt((short)(n))
#else
#define WAIT_TENSOR(n) asm volatile("s_wait_tensorcnt %0" ::"i"(n) : "memory")
#endif

static __device__ __forceinline__ v8f zero8() {
  v8f z = {0.f, 0.f, 0.f, 0.f, 0.f, 0.f, 0.f, 0.f};
  return z;
}

static __device__ __forceinline__ v8f wmma_f16(v16h a, v16h b, v8f c) {
  return __builtin_amdgcn_wmma_f32_16x16x32_f16(false, a, false, b, (short)0, c,
                                                false, false);
}

// 16x32 f16 A-fragment (ISA 7.12.2): lane L: row = L&15, g = L>>4;
// VGPR v<4 holds K = {2v,2v+1}+8g ; v>=4 holds K = {16+2(v-4),...}+8g
static __device__ __forceinline__ v16h load_a_frag(const _Float16* base, int ld,
                                                   int lane) {
  const int row = lane & 15, g = lane >> 4;
  const _Float16* p = base + row * ld + 8 * g;
  union { unsigned u[8]; v16h v; } r;
  r.u[0] = *(const unsigned*)(p + 0);
  r.u[1] = *(const unsigned*)(p + 2);
  r.u[2] = *(const unsigned*)(p + 4);
  r.u[3] = *(const unsigned*)(p + 6);
  r.u[4] = *(const unsigned*)(p + 16);
  r.u[5] = *(const unsigned*)(p + 18);
  r.u[6] = *(const unsigned*)(p + 20);
  r.u[7] = *(const unsigned*)(p + 22);
  return r.v;
}

// 32x16 f16 B-fragment from BT[n][k] storage: lane L: col=L&15, g=L>>4;
// VGPR v<4 holds K = {2v,2v+1}+8g ; v>=4 holds K = {16+2(v-4),...}+8g
static __device__ __forceinline__ v16h load_b_frag(const _Float16* bt, int ld,
                                                   int lane) {
  const int col = lane & 15, g = lane >> 4;
  const _Float16* p = bt + col * ld + 8 * g;
  union { unsigned u[8]; v16h v; } r;
#pragma unroll
  for (int v = 0; v < 8; ++v) r.u[v] = *(const unsigned*)(p + 2 * v + (v >= 4 ? 8 : 0));
  return r.v;
}

#if ATHENA_TDM
// TDM load of a 2D f16 tile (tile_w x tile_h, row pitch `stride` halves) from
// global into LDS, with the D# pad feature padding each LDS row to the bank-
// friendly stride. D# packing per CDNA5 ISA 8.3-8.5.
static __device__ __forceinline__ void tdm_tile_f16(
    const _Float16* lds_dst, const _Float16* gsrc, unsigned tile_w,
    unsigned tile_h, unsigned stride, unsigned pad_int, unsigned pad_amt) {
  const unsigned long long ga = (unsigned long long)(size_t)gsrc;
  u32x4 g0 = {1u,                               // count=1, no gather
              (unsigned)(size_t)lds_dst,        // lds_addr
              (unsigned)ga,                     // global_addr[31:0]
              (unsigned)(((ga >> 32) & 0x01FFFFFFull) | 0x80000000u)};  // type=2
  const unsigned td1 = 1u << 20;  // generous dim1 extent (always in-bounds)
  i32x8 g1 = {
      (int)((1u << 16) | (1u << 20) | (pad_int << 22) | (pad_amt << 25)),
      (int)((stride & 0xFFFFu) << 16),          // tensor_dim0 lo16
      (int)((stride >> 16) | ((td1 & 0xFFFFu) << 16)),
      (int)((td1 >> 16) | (tile_w << 16)),      // tile_dim0
      (int)tile_h,                              // tile_dim1 (tile_dim2=0)
      (int)stride,                              // tensor_dim0_stride lo32
      0, 0};
  i32x4 gz = {0, 0, 0, 0};
#if __clang_major__ >= 23
  i32x8 g4 = {0, 0, 0, 0, 0, 0, 0, 0};
  __builtin_amdgcn_tensor_load_to_lds(g0, g1, gz, gz, g4, 0);
#else
  __builtin_amdgcn_tensor_load_to_lds(g0, g1, gz, gz, 0);
#endif
}
#endif

// ---------------------------------------------------------------------------
// Pre-pass: W (f32) -> Wh = f16(W) and Wt = f16(W^T), via 32x32 LDS transpose.
// ---------------------------------------------------------------------------
__global__ __launch_bounds__(256) void convert_w_kernel(
    const float* __restrict__ W, _Float16* __restrict__ Wh,
    _Float16* __restrict__ Wt) {
  __shared__ float tile[32][33];
  const int bx = blockIdx.x * 32, by = blockIdx.y * 32;
  const int tx = threadIdx.x & 31, ty = threadIdx.x >> 5;
#pragma unroll
  for (int i = 0; i < 32; i += 8) {
    const float v = W[(size_t)(by + ty + i) * 1024 + bx + tx];
    Wh[(size_t)(by + ty + i) * 1024 + bx + tx] = (_Float16)v;
    tile[ty + i][tx] = v;
  }
  __syncthreads();
#pragma unroll
  for (int i = 0; i < 32; i += 8)
    Wt[(size_t)(bx + ty + i) * 1024 + by + tx] = (_Float16)tile[tx][ty + i];
}

// ---------------------------------------------------------------------------
// 128x128-tile WMMA GEMM: C[m,n] = sum_k A[m,k] * Bsrc[n,k]
// (Bsrc is f16, row-major in n: Wh for ZT@W^T, Wt for SSA@W.)
// 8 waves; wave = 32x64 output (2 A-frags x 4 B-frags = 8 WMMA per K-step).
// Double-buffered LDS; B (and A when A_TDM) staged by the Tensor Data Mover
// one K-step ahead of the WMMA pipeline.
// ---------------------------------------------------------------------------
template <typename TA, typename TC, bool A_TDM>
__global__ __launch_bounds__(256) void wmma_gemm_kernel(
    const TA* __restrict__ A, const _Float16* __restrict__ Bsrc,
    TC* __restrict__ C, int lda, int ldb, int ldc, int K) {
  __shared__ _Float16 At[2][128][LDG_PAD];
  __shared__ _Float16 Bt[2][128][LDG_PAD];  // BT[n][k]

  const int t = threadIdx.x, lane = t & 31, wave = t >> 5;
  const int m0 = blockIdx.y * 128, n0 = blockIdx.x * 128;
  const int wm = (wave & 3) * 32, wn = (wave >> 2) * 64;
  const int nk = K / 32;

  v8f acc[2][4];
#pragma unroll
  for (int i = 0; i < 2; ++i)
#pragma unroll
    for (int j = 0; j < 4; ++j) acc[i][j] = zero8();

#if ATHENA_TDM
  if (t < 32) {  // wave 0: prologue DMA for k-tile 0
    tdm_tile_f16(&Bt[0][0][0], Bsrc + (size_t)n0 * ldb, 32, 128, ldb, 3, 3);
    if (A_TDM)
      tdm_tile_f16(&At[0][0][0],
                   (const _Float16*)(const void*)A + (size_t)m0 * lda, 32, 128,
                   lda, 3, 3);
  }
#endif

  int cur = 0;
  for (int kt = 0; kt < nk; ++kt) {
    const int k0 = kt * 32;
    __syncthreads();  // consumers of both buffers from prior iters are done
    if (!A_TDM) {     // f32 A: stage + convert by threads
      const int r = t >> 1, c = (t & 1) * 16;
      const TA* src = A + (size_t)(m0 + r) * lda + k0 + c;
#pragma unroll
      for (int i = 0; i < 16; ++i) At[cur][r][c + i] = (_Float16)src[i];
    }
#if ATHENA_TDM
    if (t < 32) {
      if (kt + 1 < nk) {  // prefetch next k-tile into the other buffer
        tdm_tile_f16(&Bt[cur ^ 1][0][0], Bsrc + (size_t)n0 * ldb + k0 + 32, 32,
                     128, ldb, 3, 3);
        if (A_TDM)
          tdm_tile_f16(&At[cur ^ 1][0][0],
                       (const _Float16*)(const void*)A + (size_t)m0 * lda + k0 +
                           32, 32, 128, lda, 3, 3);
        WAIT_TENSOR(A_TDM ? 2 : 1);  // in-order: current tile(s) complete
      } else {
        WAIT_TENSOR(0);
      }
    }
#else
    {  // fallback synchronous staging (f16 row-major source)
      const int n = t >> 1, kk = (t & 1) * 16;
      const _Float16* src = Bsrc + (size_t)(n0 + n) * ldb + k0 + kk;
#pragma unroll
      for (int i = 0; i < 16; ++i) Bt[cur][n][kk + i] = src[i];
      if (A_TDM) {
        const int r = t >> 1, c = (t & 1) * 16;
        const _Float16* asrc =
            (const _Float16*)(const void*)A + (size_t)(m0 + r) * lda + k0 + c;
#pragma unroll
        for (int i = 0; i < 16; ++i) At[cur][r][c + i] = asrc[i];
      }
    }
#endif
    __syncthreads();  // publish At[cur], Bt[cur]

    v16h a0 = load_a_frag(&At[cur][wm][0],      LDG_PAD, lane);
    v16h a1 = load_a_frag(&At[cur][wm + 16][0], LDG_PAD, lane);
#pragma unroll
    for (int ni = 0; ni < 4; ++ni) {
      v16h bn = load_b_frag(&Bt[cur][wn + ni * 16][0], LDG_PAD, lane);
      acc[0][ni] = wmma_f16(a0, bn, acc[0][ni]);
      acc[1][ni] = wmma_f16(a1, bn, acc[1][ni]);
    }
    cur ^= 1;
  }

  const int g = lane >> 4, col = lane & 15;
#pragma unroll
  for (int mi = 0; mi < 2; ++mi)
#pragma unroll
    for (int ni = 0; ni < 4; ++ni)
#pragma unroll
      for (int v = 0; v < 8; ++v) {
        const int r = m0 + wm + mi * 16 + v + 8 * g;
        C[(size_t)r * ldc + n0 + wn + ni * 16 + col] = (TC)acc[mi][ni][v];
      }
}

// ---------------------------------------------------------------------------
// Flash attention (Q=K=V=X[b,h] of [2048,64] f16, row stride 1024).
// 8 waves x 16 query rows; 32 key tiles of 64. K tile fetched by TDM with the
// D# pad feature directly into the padded [64][72] LDS layout, double-buffered
// one tile ahead; V^T built LDS->LDS after the tensor wait.
// ---------------------------------------------------------------------------
__global__ __launch_bounds__(256) void flash_attn_kernel(
    const _Float16* __restrict__ X, _Float16* __restrict__ O) {
  __shared__ _Float16 Kt[2][64][LDF_PAD];   // BT[n=key][k=d]
  __shared__ _Float16 Vt[64][LDF_PAD];      // BT[n=d][k=key]
  __shared__ _Float16 Pb[8][16][LDF_PAD];   // per-wave P scratch

  const int t = threadIdx.x, lane = t & 31, wave = t >> 5;
  const int h = blockIdx.y, b = blockIdx.z;
  const int q0 = blockIdx.x * 128 + wave * 16;
  const int g = lane >> 4, col = lane & 15;
  const _Float16* Xbh = X + (size_t)b * 2048 * 1024 + h * 64;

#if ATHENA_TDM
  if (t < 32)  // prologue DMA for key tile 0 (overlaps with Q loads below)
    tdm_tile_f16(&Kt[0][0][0], Xbh, 64, 64, 1024, 4, 3);
#endif

  v16h aQ0 = load_a_frag(Xbh + (size_t)q0 * 1024,      1024, lane);
  v16h aQ1 = load_a_frag(Xbh + (size_t)q0 * 1024 + 32, 1024, lane);
#pragma unroll
  for (int i = 0; i < 16; ++i) {
    aQ0[i] = aQ0[i] * (_Float16)0.125f;  // DH^-0.5
    aQ1[i] = aQ1[i] * (_Float16)0.125f;
  }

  v8f o[4];
  float mrow[8], lrow[8];
#pragma unroll
  for (int i = 0; i < 4; ++i) o[i] = zero8();
#pragma unroll
  for (int v = 0; v < 8; ++v) { mrow[v] = -3.0e38f; lrow[v] = 0.f; }

  int cur = 0;
  for (int kt = 0; kt < 32; ++kt) {
    __syncthreads();  // all reads of Kt[cur^1], Vt from prev iter done
#if ATHENA_TDM
    if (t < 32) {
      if (kt + 1 < 32) {
        tdm_tile_f16(&Kt[cur ^ 1][0][0],
                     Xbh + (size_t)(kt + 1) * 64 * 1024, 64, 64, 1024, 4, 3);
        WAIT_TENSOR(1);  // in-order: Kt[cur] complete
      } else {
        WAIT_TENSOR(0);
      }
    }
    __syncthreads();  // Kt[cur] visible; build V^T from it
    {
      const int r = t >> 2, c = (t & 3) * 16;
      _Float16 tmp[16];
#pragma unroll
      for (int i = 0; i < 16; ++i) tmp[i] = Kt[cur][r][c + i];
#pragma unroll
      for (int i = 0; i < 16; ++i) Vt[c + i][r] = tmp[i];
    }
#else
    {  // fallback: synchronous K/V staging from global
      const int r = t >> 2, c = (t & 3) * 16;
      const _Float16* src = Xbh + (size_t)(kt * 64 + r) * 1024 + c;
      _Float16 tmp[16];
#pragma unroll
      for (int i = 0; i < 16; ++i) tmp[i] = src[i];
#pragma unroll
      for (int i = 0; i < 16; ++i) Kt[cur][r][c + i] = tmp[i];
#pragma unroll
      for (int i = 0; i < 16; ++i) Vt[c + i][r] = tmp[i];
    }
#endif
    __syncthreads();

    // S = (Q*scale) @ K^T : 4 key-subtiles x 2 d-steps
    v8f s[4];
#pragma unroll
    for (int nt = 0; nt < 4; ++nt) {
      v16h bk0 = load_b_frag(&Kt[cur][nt * 16][0],  LDF_PAD, lane);
      v16h bk1 = load_b_frag(&Kt[cur][nt * 16][32], LDF_PAD, lane);
      v8f sa = zero8();
      sa = wmma_f16(aQ0, bk0, sa);
      sa = wmma_f16(aQ1, bk1, sa);
      s[nt] = sa;
    }

    // --- online softmax, butterfly batched across all 8 rows per level ---
    float red[8], tmp[8];
#pragma unroll
    for (int v = 0; v < 8; ++v)
      red[v] = fmaxf(fmaxf(s[0][v], s[1][v]), fmaxf(s[2][v], s[3][v]));
#pragma unroll
    for (int off = 1; off < 16; off <<= 1) {  // 8 independent shuffles/level
#pragma unroll
      for (int v = 0; v < 8; ++v) tmp[v] = __shfl_xor(red[v], off, 32);
#pragma unroll
      for (int v = 0; v < 8; ++v) red[v] = fmaxf(red[v], tmp[v]);
    }
    float alpha[8];
#pragma unroll
    for (int v = 0; v < 8; ++v) {
      const float nm = fmaxf(mrow[v], red[v]);
      alpha[v] = __expf(mrow[v] - nm);
      mrow[v] = nm;
    }
    float rs[8];
#pragma unroll
    for (int v = 0; v < 8; ++v) {
      float r0 = 0.f;
#pragma unroll
      for (int nt = 0; nt < 4; ++nt) {
        const float p = __expf(s[nt][v] - mrow[v]);
        s[nt][v] = p;
        r0 += p;
      }
      rs[v] = r0;
    }
#pragma unroll
    for (int off = 1; off < 16; off <<= 1) {
#pragma unroll
      for (int v = 0; v < 8; ++v) tmp[v] = __shfl_xor(rs[v], off, 32);
#pragma unroll
      for (int v = 0; v < 8; ++v) rs[v] += tmp[v];
    }
#pragma unroll
    for (int v = 0; v < 8; ++v) {
      lrow[v] = lrow[v] * alpha[v] + rs[v];
#pragma unroll
      for (int nt = 0; nt < 4; ++nt) o[nt][v] = o[nt][v] * alpha[v];
    }

    // P (C/D layout) -> wave-private LDS -> A-fragment layout
    _Float16* Pw = &Pb[wave][0][0];
#pragma unroll
    for (int v = 0; v < 8; ++v) {
      _Float16* pr = Pw + (v + 8 * g) * LDF_PAD;
      pr[col]      = (_Float16)s[0][v];
      pr[16 + col] = (_Float16)s[1][v];
      pr[32 + col] = (_Float16)s[2][v];
      pr[48 + col] = (_Float16)s[3][v];
    }
    asm volatile("s_wait_dscnt 0" ::: "memory");  // same-wave LDS RAW
    v16h aP0 = load_a_frag(Pw,      LDF_PAD, lane);
    v16h aP1 = load_a_frag(Pw + 32, LDF_PAD, lane);

    // O += P @ V : 4 d-subtiles x 2 key-steps
#pragma unroll
    for (int nt = 0; nt < 4; ++nt) {
      v16h bv0 = load_b_frag(&Vt[nt * 16][0],  LDF_PAD, lane);
      v16h bv1 = load_b_frag(&Vt[nt * 16][32], LDF_PAD, lane);
      o[nt] = wmma_f16(aP0, bv0, o[nt]);
      o[nt] = wmma_f16(aP1, bv1, o[nt]);
    }
    cur ^= 1;
  }

  // normalize and write SSA (f16) in [b, n, h*64+d] layout
#pragma unroll
  for (int v = 0; v < 8; ++v) {
    const float inv = 1.f / lrow[v];
    _Float16* dst = O + ((size_t)b * 2048 + q0 + v + 8 * g) * 1024 + h * 64;
#pragma unroll
    for (int nt = 0; nt < 4; ++nt)
      dst[nt * 16 + col] = (_Float16)(o[nt][v] * inv);
  }
}

// ---------------------------------------------------------------------------
extern "C" void kernel_launch(void* const* d_in, const int* in_sizes, int n_in,
                              void* d_out, int out_size, void* d_ws,
                              size_t ws_size, hipStream_t stream) {
  const float* ZT = (const float*)d_in[0];  // [4,2048,1024] f32
  const float* W  = (const float*)d_in[1];  // [1024,1024]  f32
  float* OUT = (float*)d_out;               // [4,2048,1024] f32

  _Float16* ZTU = (_Float16*)d_ws;                    // 16 MB
  _Float16* SSA = ZTU + (size_t)8192 * 1024;          // 16 MB
  _Float16* Wh  = SSA + (size_t)8192 * 1024;          // 2 MB, f16(W)
  _Float16* Wt  = Wh + (size_t)1024 * 1024;           // 2 MB, f16(W^T)

  // 0) one-time W conversion (+transpose)
  convert_w_kernel<<<dim3(32, 32), 256, 0, stream>>>(W, Wh, Wt);
  // 1) ZTU = f16( ZT @ W^T )  -> B tiles from Wh rows (TDM)
  wmma_gemm_kernel<float, _Float16, false>
      <<<dim3(1024 / 128, 8192 / 128), 256, 0, stream>>>(ZT, Wh, ZTU, 1024,
                                                         1024, 1024, 1024);
  // 2) SSA = flash_attention(ZTU) per (b,h)
  flash_attn_kernel<<<dim3(2048 / 128, 16, 4), 256, 0, stream>>>(ZTU, SSA);
  // 3) OUT = SSA @ W  -> A tiles from SSA (TDM), B tiles from Wt rows (TDM)
  wmma_gemm_kernel<_Float16, float, true>
      <<<dim3(1024 / 128, 8192 / 128), 256, 0, stream>>>(SSA, Wt, OUT, 1024,
                                                         1024, 1024, 1024);
}
